// SNN_61890478735918
// MI455X (gfx1250) — matrix operand order repaired and checked by reading, not compiled, not run on hardware
//
#include <hip/hip_runtime.h>

typedef __attribute__((ext_vector_type(16))) _Float16 v16h;
typedef __attribute__((ext_vector_type(8)))  float    v8f;
typedef __attribute__((ext_vector_type(4)))  unsigned int v4u;
typedef __attribute__((ext_vector_type(8)))  int      v8i;
typedef __attribute__((ext_vector_type(4)))  int      v4i;

// Problem constants (from reference)
constexpr int Tn  = 128;
constexpr int Bn  = 256;
constexpr int D1n = 784;
constexpr int D2n = 512;

// LIF constants
constexpr float ALPHA  = 0.818730753f;   // exp(-0.001/0.005)
constexpr float BETA   = 0.951229425f;   // exp(-0.001/0.02)
constexpr float OMBETA = 1.0f - BETA;

// Tiling
constexpr int BT    = 16;     // batch rows per workgroup
constexpr int OT    = 64;     // output cols per workgroup (4 waves x 16)
constexpr int PITCH = 808;    // LDS row pitch in halves (bank-conflict-free)
constexpr int NTHR  = 128;    // 4 waves

// Issue a TDM load of one [16 x 784] f32 tile (row stride D1) into LDS.
// Descriptor layout per CDNA5 ISA ch.8 (groups 0/1; groups 2/3 zero for 2-D).
// This toolchain exposes the 6-arg builtin:
//   (uint32x4 g0, int32x8 g1, int32x4 g2, int32x4 g3, int32x8 gx, i32 cpol)
__device__ __forceinline__ void tdm_load_x_tile(const float* gsrc,
                                                unsigned lds_byte_addr) {
  const unsigned long long ga = (unsigned long long)(const void*)gsrc;
  v4u g0;
  g0.x = 1u;                                   // count=1, user descriptor
  g0.y = lds_byte_addr;                        // lds_addr [63:32]
  g0.z = (unsigned)(ga & 0xFFFFFFFFull);       // global_addr[31:0]
  g0.w = (unsigned)((ga >> 32) & 0x01FFFFFFull) | (2u << 30); // addr[56:32], type=2
  v8i g1;
  g1[0] = 0x00020000;                          // data_size=2 (4 bytes/elem)
  g1[1] = (int)(784u << 16);                   // tensor_dim0 = 784 (low16 at [63:48])
  g1[2] = (int)(16u << 16);                    // tensor_dim0 hi=0; tensor_dim1 = 16
  g1[3] = (int)(784u << 16);                   // tensor_dim1 hi=0; tile_dim0 = 784
  g1[4] = 16;                                  // tile_dim1 = 16; tile_dim2 = 0
  g1[5] = 784;                                 // tensor_dim0_stride low32 = 784
  g1[6] = 0;                                   // stride hi; tensor_dim1_stride lo
  g1[7] = 0;
  const v4i gz4 = {0, 0, 0, 0};
  const v8i gz8 = {0, 0, 0, 0, 0, 0, 0, 0};
  __builtin_amdgcn_tensor_load_to_lds(g0, g1, gz4, gz4, gz8, 0);
}

__global__ void snn_fused_wmma(const float* __restrict__ x,
                               const float* __restrict__ W,
                               float* __restrict__ out) {
  extern __shared__ char smem[];
  _Float16* wlds = (_Float16*)smem;                    // [OT][PITCH] f16
  _Float16* xlds = wlds + OT * PITCH;                  // [BT][PITCH] f16
  float*    xraw = (float*)(smem + (size_t)(OT + BT) * PITCH * 2); // [BT][784] f32

  const int tid  = threadIdx.x;
  const int lane = tid & 31;
  const int wave = tid >> 5;
  const int n    = lane & 15;          // column within 16x16 tile
  const int hi   = lane >> 4;          // lane half select
  const int b0   = blockIdx.x * BT;
  const int o0   = blockIdx.y * OT;

  const size_t BD1 = (size_t)Bn * D1n;
  const size_t BD2 = (size_t)Bn * D2n;
  const float* xtile0 = x + (size_t)b0 * D1n;          // x[0] tile for this WG
  const unsigned xraw_lds = (unsigned)(unsigned long long)(void*)xraw;

  // Kick off TDM for x[0] immediately; it overlaps the W staging below.
  if (wave == 0) tdm_load_x_tile(xtile0, xraw_lds);

  // ---- stage W[o0:o0+OT, :] into LDS as f16 (once) ----
  for (int idx = tid; idx < OT * (D1n / 4); idx += NTHR) {
    const int r  = idx / (D1n / 4);
    const int kc = (idx % (D1n / 4)) * 4;
    const float4 f = *(const float4*)(W + (size_t)(o0 + r) * D1n + kc);
    union { _Float16 h[4]; uint2 u; } p;
    p.h[0] = (_Float16)f.x; p.h[1] = (_Float16)f.y;
    p.h[2] = (_Float16)f.z; p.h[3] = (_Float16)f.w;
    *(uint2*)(wlds + r * PITCH + kc) = p.u;
  }
  // zero K-pad region of W and x f16 buffers (written once, never dirtied)
  for (int idx = tid; idx < OT * (PITCH - D1n); idx += NTHR) {
    const int r = idx / (PITCH - D1n);
    const int c = idx % (PITCH - D1n);
    wlds[r * PITCH + D1n + c] = (_Float16)0.0f;
  }
  for (int idx = tid; idx < BT * (PITCH - D1n); idx += NTHR) {
    const int r = idx / (PITCH - D1n);
    const int c = idx % (PITCH - D1n);
    xlds[r * PITCH + D1n + c] = (_Float16)0.0f;
  }
  // out[0] is all zeros in the reference
  for (int idx = tid; idx < BT * OT; idx += NTHR) {
    const int r = idx / OT, c = idx % OT;
    out[(size_t)(b0 + r) * D2n + (o0 + c)] = 0.0f;
  }

  // LIF state lives in accumulator layout (VGPR j <-> row hi*8+j, col n)
  float I[8], V[8];
#pragma unroll
  for (int j = 0; j < 8; ++j) { I[j] = 0.0f; V[j] = 0.0f; }

  // Per-lane fragment base pointers (16-bit A 16x32 / B 32x16, contiguous k = 16*hi + s)
  const _Float16* abase = xlds + n * PITCH;                 // row = lane%16
  const _Float16* bbase = wlds + (wave * 16 + n) * PITCH;   // col = lane%16
  const int a_off = hi * 16;    // lane>=16: K starts +16 within chunk
  const int b_off = hi * 16;    // lane>=16: K starts +16 within chunk
  const int ocol  = o0 + wave * 16 + n;

#pragma unroll 1
  for (int t = 0; t < Tn - 1; ++t) {
    // x[t] raw tile must have landed in LDS (TDM issued last iteration /
    // prologue). Only wave0's TENSORcnt tracks it; publish via barrier.
    if (wave == 0) __builtin_amdgcn_s_wait_tensorcnt(0);
    __syncthreads();

    // ---- convert raw f32 x[t] -> padded f16 staging (pure LDS traffic) ----
    for (int idx = tid; idx < BT * (D1n / 4); idx += NTHR) {
      const int r  = idx / (D1n / 4);
      const int kc = (idx % (D1n / 4)) * 4;
      const float4 f = *(const float4*)(xraw + r * D1n + kc);
      union { _Float16 h[4]; uint2 u; } p;
      p.h[0] = (_Float16)f.x; p.h[1] = (_Float16)f.y;
      p.h[2] = (_Float16)f.z; p.h[3] = (_Float16)f.w;
      *(uint2*)(xlds + r * PITCH + kc) = p.u;
    }
    __syncthreads();   // conversion done: xraw free, xlds(t) ready

    // Prefetch x[t+1] via the Tensor Data Mover; overlaps the WMMAs below.
    if (wave == 0 && (t + 1) < (Tn - 1))
      tdm_load_x_tile(x + (size_t)(t + 1) * BD1 + (size_t)b0 * D1n, xraw_lds);

    // ---- drive tile: 25 x v_wmma_f32_16x16x32_f16, 2 accumulators ----
    v8f acc0 = {0.f, 0.f, 0.f, 0.f, 0.f, 0.f, 0.f, 0.f};
    v8f acc1 = {0.f, 0.f, 0.f, 0.f, 0.f, 0.f, 0.f, 0.f};
    for (int kb = 0; kb < 768; kb += 64) {
      union { v16h v; uint4 q[2]; } a0, b0f, a1, b1f;
      a0.q[0]  = *(const uint4*)(abase + kb + a_off);
      a0.q[1]  = *(const uint4*)(abase + kb + a_off + 8);
      b0f.q[0] = *(const uint4*)(bbase + kb + b_off);
      b0f.q[1] = *(const uint4*)(bbase + kb + b_off + 8);
      a1.q[0]  = *(const uint4*)(abase + kb + 32 + a_off);
      a1.q[1]  = *(const uint4*)(abase + kb + 32 + a_off + 8);
      b1f.q[0] = *(const uint4*)(bbase + kb + 32 + b_off);
      b1f.q[1] = *(const uint4*)(bbase + kb + 32 + b_off + 8);
      acc0 = __builtin_amdgcn_wmma_f32_16x16x32_f16(
          false, a0.v, false, b0f.v, (short)0, acc0, false, false);
      acc1 = __builtin_amdgcn_wmma_f32_16x16x32_f16(
          false, a1.v, false, b1f.v, (short)0, acc1, false, false);
    }
    {  // final chunk kb = 768 (covers K 768..799, padded region zeroed)
      union { v16h v; uint4 q[2]; } a, b;
      a.q[0] = *(const uint4*)(abase + 768 + a_off);
      a.q[1] = *(const uint4*)(abase + 768 + a_off + 8);
      b.q[0] = *(const uint4*)(bbase + 768 + b_off);
      b.q[1] = *(const uint4*)(bbase + 768 + b_off + 8);
      acc0 = __builtin_amdgcn_wmma_f32_16x16x32_f16(
          false, a.v, false, b.v, (short)0, acc0, false, false);
    }
    const v8f acc = acc0 + acc1;

    // ---- LIF update + spike store (row = b0 + hi*8 + j, col = ocol) ----
    float* orow = out + (size_t)(t + 1) * BD2 + ocol;
#pragma unroll
    for (int j = 0; j < 8; ++j) {
      I[j] = ALPHA * I[j] + acc[j];
      const float vpre = BETA * V[j] + OMBETA * I[j];
      const float s = (vpre > 1.0f) ? 1.0f : 0.0f;
      V[j] = (1.0f - s) * vpre;   // reset-to-zero
      orow[(size_t)(b0 + hi * 8 + j) * D2n] = s;
    }
  }
}

extern "C" void kernel_launch(void* const* d_in, const int* in_sizes, int n_in,
                              void* d_out, int out_size, void* d_ws, size_t ws_size,
                              hipStream_t stream) {
  (void)in_sizes; (void)n_in; (void)out_size; (void)d_ws; (void)ws_size;
  const float* x = (const float*)d_in[0];   // [T, B, D1] f32 (binary)
  const float* W = (const float*)d_in[1];   // [D2, D1]  f32
  float* out = (float*)d_out;               // [T, B, D2] f32

  dim3 grid(Bn / BT, D2n / OT);   // 16 x 8 = 128 workgroups
  dim3 block(NTHR);               // 4 waves
  const size_t shmem = (size_t)(OT + BT) * PITCH * sizeof(_Float16)
                     + (size_t)BT * D1n * sizeof(float);   // ~176 KB < 320 KB/WGP
  hipLaunchKernelGGL(snn_fused_wmma, grid, block, shmem, stream, x, W, out);
}
